// MultiHeadSelfAttention_26259430047749
// MI455X (gfx1250) — matrix operand order repaired and checked by reading, not compiled, not run on hardware
//
#include <hip/hip_runtime.h>

// ---------------- problem dims ----------------
#define EMBED 1024
#define SEQ   2048
#define NHEAD 16
#define DHEAD 64
#define BATCH 2
#define ROWS  (BATCH * SEQ)   // 4096

// ---------------- WMMA fragment types ----------------
typedef __attribute__((ext_vector_type(16))) __bf16 v16bf;
typedef __attribute__((ext_vector_type(8)))  __bf16 bf16x8;
typedef __attribute__((ext_vector_type(8)))  float  v8f;
typedef __attribute__((ext_vector_type(4)))  unsigned int u32x4;
typedef __attribute__((ext_vector_type(8)))  int i32x8;
typedef __attribute__((ext_vector_type(4)))  int i32x4;

union FragU { v16bf v; bf16x8 h[2]; };

// A-matrix 16x32 bf16 fragment (row-major source, rows contiguous along K).
// lanes 0-15: row = lane, K in {0..7, 16..23}; lanes 16-31: row = lane-16, K in {8..15, 24..31}
__device__ __forceinline__ v16bf load_a_frag(const __bf16* base, int stride, int lane) {
    int r    = lane & 15;
    int koff = (lane >> 4) << 3;   // 0 or 8
    const __bf16* p = base + r * stride + koff;
    FragU f;
    f.h[0] = *(const bf16x8*)(p);        // K = koff .. koff+7
    f.h[1] = *(const bf16x8*)(p + 16);   // K = koff+16 .. koff+23
    return f.v;
}

// B-matrix 32x16 bf16 fragment: column n contiguous along K in memory.
// lanes 0-15: col = lane, K in {0..7, 16..23}; lanes 16-31: col = lane-16, K in {8..15, 24..31}
__device__ __forceinline__ v16bf load_b_frag(const __bf16* base, int stride, int lane) {
    int n    = lane & 15;
    int koff = (lane >> 4) << 3;   // 0 or 8
    const __bf16* p = base + n * stride + koff;
    FragU f;
    f.h[0] = *(const bf16x8*)(p);
    f.h[1] = *(const bf16x8*)(p + 16);
    return f.v;
}

__device__ __forceinline__ float redmax16(float v) {
    v = fmaxf(v, __shfl_xor(v, 1));
    v = fmaxf(v, __shfl_xor(v, 2));
    v = fmaxf(v, __shfl_xor(v, 4));
    v = fmaxf(v, __shfl_xor(v, 8));
    return v;
}
__device__ __forceinline__ float redsum16(float v) {
    v += __shfl_xor(v, 1);
    v += __shfl_xor(v, 2);
    v += __shfl_xor(v, 4);
    v += __shfl_xor(v, 8);
    return v;
}

// ---------------- TDM: 2-D tile DMA global -> LDS ----------------
// D# layout per CDNA5 ISA. Tile fully in-bounds, so
// tensor dims are set equal to tile dims. data_size = 1 (2-byte bf16 elems).
__device__ __forceinline__ unsigned lds_offset(const void* p) {
    // LDS aperture: flat address low 32 bits == LDS byte offset (ISA §10.2)
    return (unsigned)(unsigned long long)(uintptr_t)p;
}

__device__ __forceinline__ void tdm_load_2d(const void* gptr, unsigned lds_off,
                                            unsigned tile_x, unsigned tile_y,
                                            unsigned stride_x /*elems*/) {
    unsigned long long ga = (unsigned long long)(uintptr_t)gptr;
    u32x4 g0;
    g0[0] = 1u;                                   // count=1, user descriptor
    g0[1] = lds_off;                              // lds_addr (bytes)
    g0[2] = (unsigned)(ga & 0xffffffffu);         // global_addr[31:0]
    g0[3] = (unsigned)((ga >> 32) & 0x01ffffffu)  // global_addr[56:32]
          | (2u << 30);                           // type = 2 ("image")
    i32x8 g1;
    g1[0] = (int)(1u << 16);                      // data_size = 1 (2 bytes)
    g1[1] = (int)((tile_x & 0xffffu) << 16);      // tensor_dim0[15:0]
    g1[2] = (int)((tile_x >> 16) | ((tile_y & 0xffffu) << 16)); // dim0[31:16], dim1[15:0]
    g1[3] = (int)((tile_y >> 16) | ((tile_x & 0xffffu) << 16)); // dim1[31:16], tile_dim0
    g1[4] = (int)(tile_y & 0xffffu);              // tile_dim1 (tile_dim2 = 0)
    g1[5] = (int)stride_x;                        // tensor_dim0_stride[31:0]
    g1[6] = 0;                                    // stride[47:32], dim1_stride[15:0]
    g1[7] = 0;
    i32x4 z4 = {0, 0, 0, 0};
#if defined(__clang_major__) && (__clang_major__ >= 23)
    i32x8 z8 = {0, 0, 0, 0, 0, 0, 0, 0};
    __builtin_amdgcn_tensor_load_to_lds(g0, g1, z4, z4, z8, 0);
#else
    __builtin_amdgcn_tensor_load_to_lds(g0, g1, z4, z4, 0);
#endif
}

// ---------------- kernel 1: f32 -> bf16 ----------------
__global__ void k_cvt_bf16(const float* __restrict__ in, __bf16* __restrict__ out, int n) {
    int i = blockIdx.x * blockDim.x + threadIdx.x;
    if (i < n) out[i] = (__bf16)in[i];
}

// ---------------- kernel 2: C = A * W^T + bias  (TDM double-buffered, bf16 WMMA) ----------------
// A: [M,K] bf16 row-major.  W: [N,K] bf16 row-major (nn.Linear weight).
// Block = 8 waves -> 128x256 tile; wave = 64x64 (4x4 WMMA tiles).
// Wave 0 drives the TDM: per k-step DMA A(128x32) and W(256x32) tiles into LDS,
// double-buffered on TENSORcnt, all waves consume fragments from LDS.
__global__ __launch_bounds__(256)
void k_gemm_bf16(const __bf16* __restrict__ A, const __bf16* __restrict__ W,
                 const float* __restrict__ bias,
                 __bf16* __restrict__ outB, float* __restrict__ outF,
                 int M, int N, int K) {
    __shared__ __bf16 sA[2][128 * 32];   // 8 KB per stage
    __shared__ __bf16 sB[2][256 * 32];   // 16 KB per stage

    const int lane  = threadIdx.x & 31;
    const int wave  = threadIdx.x >> 5;
    const int waveM = wave >> 2;         // 0..1
    const int waveN = wave & 3;          // 0..3
    const int m0    = blockIdx.x * 128;
    const int n0    = blockIdx.y * 256;
    const int NS    = K / 32;

    const v8f vzero = {0.f,0.f,0.f,0.f,0.f,0.f,0.f,0.f};
    v8f acc[4][4];
#pragma unroll
    for (int i = 0; i < 4; ++i)
#pragma unroll
        for (int j = 0; j < 4; ++j) acc[i][j] = vzero;

    if (wave == 0) {
        tdm_load_2d(A + (size_t)m0 * K, lds_offset(sA[0]), 32, 128, (unsigned)K);
        tdm_load_2d(W + (size_t)n0 * K, lds_offset(sB[0]), 32, 256, (unsigned)K);
    }

    for (int kk = 0; kk < NS; ++kk) {
        const int cur = kk & 1;
        if (wave == 0) {
            if (kk + 1 < NS) {
                tdm_load_2d(A + (size_t)m0 * K + (kk + 1) * 32,
                            lds_offset(sA[cur ^ 1]), 32, 128, (unsigned)K);
                tdm_load_2d(W + (size_t)n0 * K + (kk + 1) * 32,
                            lds_offset(sB[cur ^ 1]), 32, 256, (unsigned)K);
                __builtin_amdgcn_s_wait_tensorcnt(2);   // stage kk complete
            } else {
                __builtin_amdgcn_s_wait_tensorcnt(0);
            }
        }
        __syncthreads();                 // publish stage kk tile to all waves

        v16bf a[4], b[4];
#pragma unroll
        for (int mt = 0; mt < 4; ++mt)
            a[mt] = load_a_frag(sA[cur] + (waveM * 64 + mt * 16) * 32, 32, lane);
#pragma unroll
        for (int nt = 0; nt < 4; ++nt)
            b[nt] = load_b_frag(sB[cur] + (waveN * 64 + nt * 16) * 32, 32, lane);
#pragma unroll
        for (int mt = 0; mt < 4; ++mt)
#pragma unroll
            for (int nt = 0; nt < 4; ++nt)
                acc[mt][nt] = __builtin_amdgcn_wmma_f32_16x16x32_bf16(
                    false, a[mt], false, b[nt], (short)0, acc[mt][nt], false, false);

        __syncthreads();                 // all waves done before buffer reuse
    }

    const int hi = lane >> 4, nn = lane & 15;
#pragma unroll
    for (int mt = 0; mt < 4; ++mt)
#pragma unroll
        for (int nt = 0; nt < 4; ++nt)
#pragma unroll
            for (int r = 0; r < 8; ++r) {
                int row = m0 + waveM * 64 + mt * 16 + hi * 8 + r;
                int col = n0 + waveN * 64 + nt * 16 + nn;
                float v = acc[mt][nt][r] + bias[col];
                if (outF) outF[(size_t)row * N + col] = v;
                else      outB[(size_t)row * N + col] = (__bf16)v;
            }
}

// ---------------- kernel 3: V transpose -> Vt[b][h][d][s] ----------------
__global__ void k_transpose_v(const __bf16* __restrict__ V, __bf16* __restrict__ Vt) {
    int i = blockIdx.x * blockDim.x + threadIdx.x;   // over BATCH*NHEAD*DHEAD*SEQ
    int s = i & (SEQ - 1);
    int d = (i >> 11) & 63;
    int h = (i >> 17) & 15;
    int b = i >> 21;
    Vt[i] = V[(size_t)(b * SEQ + s) * EMBED + h * DHEAD + d];
}

// ---------------- kernel 4: flash attention ----------------
// One wave handles one (batch, head, 64-query tile). Online softmax in registers,
// P staged through LDS to convert the D-matrix layout into A fragments.
__global__ __launch_bounds__(32)
void k_attention(const __bf16* __restrict__ Q, const __bf16* __restrict__ Kmat,
                 const __bf16* __restrict__ Vt, __bf16* __restrict__ Ctx) {
    __shared__ __bf16 sP[64 * 72];   // 64x64 P tile, padded stride

    const int lane = threadIdx.x;
    const int q0   = blockIdx.x * 64;
    const int h    = blockIdx.y;
    const int b    = blockIdx.z;
    const float scale = 0.125f;      // 1/sqrt(64)
    const int hi = lane >> 4, nn = lane & 15;

    // Q fragments: 64 rows x 64 dims, K-dim = d split into 2 steps of 32
    v16bf aQ[4][2];
    const __bf16* qbase = Q + (size_t)(b * SEQ + q0) * EMBED + h * DHEAD;
#pragma unroll
    for (int mt = 0; mt < 4; ++mt)
#pragma unroll
        for (int ks = 0; ks < 2; ++ks)
            aQ[mt][ks] = load_a_frag(qbase + mt * 16 * EMBED + ks * 32, EMBED, lane);

    const v8f vzero = {0.f,0.f,0.f,0.f,0.f,0.f,0.f,0.f};
    float m_i[4][8], l_i[4][8];
    v8f accO[4][4];
#pragma unroll
    for (int mt = 0; mt < 4; ++mt) {
#pragma unroll
        for (int r = 0; r < 8; ++r) { m_i[mt][r] = -1e30f; l_i[mt][r] = 0.f; }
#pragma unroll
        for (int nt = 0; nt < 4; ++nt) accO[mt][nt] = vzero;
    }

    for (int kt = 0; kt < SEQ / 64; ++kt) {
        // ---- S = Q K^T on a 64x64 tile ----
        v8f accS[4][4];
#pragma unroll
        for (int mt = 0; mt < 4; ++mt)
#pragma unroll
            for (int nt = 0; nt < 4; ++nt) accS[mt][nt] = vzero;

#pragma unroll
        for (int ks = 0; ks < 2; ++ks) {
            v16bf bK[4];
#pragma unroll
            for (int nt = 0; nt < 4; ++nt)
                bK[nt] = load_b_frag(
                    Kmat + (size_t)(b * SEQ + kt * 64 + nt * 16) * EMBED + h * DHEAD + ks * 32,
                    EMBED, lane);
#pragma unroll
            for (int mt = 0; mt < 4; ++mt)
#pragma unroll
                for (int nt = 0; nt < 4; ++nt)
                    accS[mt][nt] = __builtin_amdgcn_wmma_f32_16x16x32_bf16(
                        false, aQ[mt][ks], false, bK[nt], (short)0, accS[mt][nt], false, false);
        }

        // ---- online softmax update (row stats across 16-lane groups) ----
#pragma unroll
        for (int mt = 0; mt < 4; ++mt) {
#pragma unroll
            for (int r = 0; r < 8; ++r) {
                float mx = -1e30f;
#pragma unroll
                for (int nt = 0; nt < 4; ++nt) mx = fmaxf(mx, accS[mt][nt][r]);
                mx = redmax16(mx * scale);
                float mnew  = fmaxf(m_i[mt][r], mx);
                float alpha = __expf(m_i[mt][r] - mnew);
                float rsum  = 0.f;
#pragma unroll
                for (int nt = 0; nt < 4; ++nt) {
                    float p = __expf(accS[mt][nt][r] * scale - mnew);
                    accS[mt][nt][r] = p;
                    rsum += p;
                }
                rsum = redsum16(rsum);
                l_i[mt][r] = l_i[mt][r] * alpha + rsum;
                m_i[mt][r] = mnew;
#pragma unroll
                for (int nt = 0; nt < 4; ++nt) accO[mt][nt][r] *= alpha;
            }
            // stage P (bf16) into LDS, row-major [query][key]
#pragma unroll
            for (int nt = 0; nt < 4; ++nt)
#pragma unroll
                for (int r = 0; r < 8; ++r)
                    sP[(mt * 16 + hi * 8 + r) * 72 + nt * 16 + nn] = (__bf16)accS[mt][nt][r];
        }
        __syncthreads();

        // ---- O += P * V ----
#pragma unroll
        for (int ks = 0; ks < 2; ++ks) {
            v16bf aP[4], bV[4];
#pragma unroll
            for (int mt = 0; mt < 4; ++mt)
                aP[mt] = load_a_frag(sP + mt * 16 * 72 + ks * 32, 72, lane);
#pragma unroll
            for (int nt = 0; nt < 4; ++nt)
                bV[nt] = load_b_frag(
                    Vt + ((size_t)((b * NHEAD + h) * DHEAD + nt * 16)) * SEQ + kt * 64 + ks * 32,
                    SEQ, lane);
#pragma unroll
            for (int mt = 0; mt < 4; ++mt)
#pragma unroll
                for (int nt = 0; nt < 4; ++nt)
                    accO[mt][nt] = __builtin_amdgcn_wmma_f32_16x16x32_bf16(
                        false, aP[mt], false, bV[nt], (short)0, accO[mt][nt], false, false);
        }
        __syncthreads();
    }

    // ---- normalize and write context (bf16, [b][s][h][d] layout) ----
#pragma unroll
    for (int mt = 0; mt < 4; ++mt)
#pragma unroll
        for (int nt = 0; nt < 4; ++nt)
#pragma unroll
            for (int r = 0; r < 8; ++r) {
                int row = q0 + mt * 16 + hi * 8 + r;
                int col = h * DHEAD + nt * 16 + nn;
                Ctx[(size_t)(b * SEQ + row) * EMBED + col] =
                    (__bf16)(accO[mt][nt][r] / l_i[mt][r]);
            }
}

// ---------------- host-side launcher ----------------
extern "C" void kernel_launch(void* const* d_in, const int* in_sizes, int n_in,
                              void* d_out, int out_size, void* d_ws, size_t ws_size,
                              hipStream_t stream) {
    const float* x  = (const float*)d_in[0];
    const float* Wq = (const float*)d_in[1];
    const float* bq = (const float*)d_in[2];
    const float* Wk = (const float*)d_in[3];
    const float* bk = (const float*)d_in[4];
    const float* Wv = (const float*)d_in[5];
    const float* bv = (const float*)d_in[6];
    const float* Wo = (const float*)d_in[7];
    const float* bo = (const float*)d_in[8];

    char* ws = (char*)d_ws;
    const size_t MB = 1u << 20;
    __bf16* xb  = (__bf16*)(ws + 0 * MB);     // 8 MB  x bf16
    __bf16* wqb = (__bf16*)(ws + 8 * MB);     // 2 MB
    __bf16* wkb = (__bf16*)(ws + 10 * MB);
    __bf16* wvb = (__bf16*)(ws + 12 * MB);
    __bf16* wob = (__bf16*)(ws + 14 * MB);
    __bf16* Qb  = (__bf16*)(ws + 16 * MB);    // 8 MB each
    __bf16* Kb  = (__bf16*)(ws + 24 * MB);
    __bf16* Vb  = (__bf16*)(ws + 32 * MB);
    __bf16* Vt  = (__bf16*)(ws + 40 * MB);
    __bf16* Ctx = (__bf16*)(ws + 48 * MB);    // total 56 MB

    const int NX = ROWS * EMBED;        // 4M
    const int NW = EMBED * EMBED;       // 1M
    k_cvt_bf16<<<(NX + 255) / 256, 256, 0, stream>>>(x,  xb,  NX);
    k_cvt_bf16<<<(NW + 255) / 256, 256, 0, stream>>>(Wq, wqb, NW);
    k_cvt_bf16<<<(NW + 255) / 256, 256, 0, stream>>>(Wk, wkb, NW);
    k_cvt_bf16<<<(NW + 255) / 256, 256, 0, stream>>>(Wv, wvb, NW);
    k_cvt_bf16<<<(NW + 255) / 256, 256, 0, stream>>>(Wo, wob, NW);

    dim3 ggrid(ROWS / 128, EMBED / 256);
    k_gemm_bf16<<<ggrid, 256, 0, stream>>>(xb, wqb, bq, Qb, (float*)nullptr, ROWS, EMBED, EMBED);
    k_gemm_bf16<<<ggrid, 256, 0, stream>>>(xb, wkb, bk, Kb, (float*)nullptr, ROWS, EMBED, EMBED);
    k_gemm_bf16<<<ggrid, 256, 0, stream>>>(xb, wvb, bv, Vb, (float*)nullptr, ROWS, EMBED, EMBED);

    k_transpose_v<<<(NX + 255) / 256, 256, 0, stream>>>(Vb, Vt);

    dim3 agrid(SEQ / 64, NHEAD, BATCH);
    k_attention<<<agrid, 32, 0, stream>>>(Qb, Kb, Vt, Ctx);

    k_gemm_bf16<<<ggrid, 256, 0, stream>>>(Ctx, wob, bo, (__bf16*)nullptr, (float*)d_out,
                                           ROWS, EMBED, EMBED);
}
